// sLSTMBlock_48541720379802
// MI455X (gfx1250) — compile-verified, matrix operand order checked by reading, not run on hardware
//
#include <hip/hip_runtime.h>
#include <hip/hip_bf16.h>

// ---------------------------------------------------------------------------
// sLSTM block for MI455X (gfx1250, wave32, WMMA).
//
// Sequential scan => one persistent 1024-thread workgroup carries all 8 batch
// states; each step's GEMVs are batched into N=8 columns of
// V_WMMA_F32_16X16X32_BF16 tiles (cols 8..15 zero padding via a zeroed LDS
// region, branchless). Weights are converted fp32->bf16 once into d_ws,
// pre-swizzled into the CDNA5 16x32 A-fragment lane layout so each lane loads
// 32B contiguously. Recurrent scalars (c, m, n, conv ring) live in registers;
// activation vectors live in LDS as bf16 B-operands. ~177KB LDS.
// ---------------------------------------------------------------------------

#define BB   8
#define TT   2048
#define DD   1024
#define NH   4
#define HS   256
#define UU   1365
#define EPSF 1e-5f

typedef __attribute__((ext_vector_type(16))) __bf16       v16bf;
typedef __attribute__((ext_vector_type(8)))  float        v8f;
typedef __attribute__((ext_vector_type(8)))  unsigned int v8u;

// d_ws layout (units: ushort/bf16 elements). Tiles are (rowTile, kTile) with
// 512 bf16 per tile stored as 32 lanes x 16 elements (exact A-frag order).
#define OFF_IF   0u          // 128 x 32 tiles  (2048 x 1024)
#define OFF_ZO   2097152u    // 128 x 32 tiles  (2048 x 1024)
#define OFF_HH   4194304u    // 256 x 8  tiles  (4096 x 256, block-diag heads)
#define OFF_UP   5242880u    // 171 x 32 tiles  (2730->2736 x 1024)
#define OFF_DOWN 8044544u    // 64  x 43 tiles  (1024 x 1365->1376)
// total = 9,453,568 ushorts = ~18.9 MB of d_ws

__device__ __forceinline__ unsigned short f2bf(float f) {
  unsigned u = __builtin_bit_cast(unsigned, f);
  unsigned r = (u + 0x7FFFu + ((u >> 16) & 1u)) >> 16;   // RNE
  return (unsigned short)r;
}

// A fragment: 32 bytes contiguous per lane (pre-swizzled in d_ws).
__device__ __forceinline__ v16bf load_afrag(const unsigned short* tileBase) {
  int l = threadIdx.x & 31;
  v8u r = *(const v8u*)(tileBase + l * 16);
  return __builtin_bit_cast(v16bf, r);
}

// B fragment (32K x 16N) built from an LDS bf16 vector per batch column.
// Lane l: N = l&15 (batch; >=8 reads a zeroed LDS region -> branchless),
// half = l>>4.
// dword v<4  holds K = k0 + half*8 + 2v        , +1
// dword v>=4 holds K = k0 + 16 + half*8 + 2(v-4), +1
__device__ __forceinline__ v16bf build_bfrag(const unsigned short* base,
                                             int vecStride, int k0,
                                             const unsigned short* zero) {
  int l = threadIdx.x & 31;
  int n = l & 15, half = l >> 4;
  const unsigned short* p =
      (n < 8) ? (base + n * vecStride + k0 + half * 8) : zero;
  v8u r;
#pragma unroll
  for (int v = 0; v < 8; ++v) {
    int ko = (v < 4) ? (2 * v) : (16 + 2 * (v - 4));
    r[v] = *(const unsigned int*)(p + ko);
  }
  return __builtin_bit_cast(v16bf, r);
}

#define WMMA_BF16(A, B, C) \
  __builtin_amdgcn_wmma_f32_16x16x32_bf16(false, (A), false, (B), (short)0, (C), false, false)

// ---------------------------------------------------------------------------
// Pre-kernel: fp32 -> bf16 with WMMA A-fragment swizzle, zero padding OOB.
// ---------------------------------------------------------------------------
__global__ void pack_weights_kernel(const float* __restrict__ wif,
                                    const float* __restrict__ wzo,
                                    const float* __restrict__ whh,
                                    const float* __restrict__ wup,
                                    const float* __restrict__ wdn,
                                    unsigned short* __restrict__ ws) {
  int m = blockIdx.y;
  const float* src; unsigned off; int RT, KT, R, K;
  switch (m) {
    case 0:  src = wif; off = OFF_IF;   RT = 128; KT = 32; R = 2048; K = 1024; break;
    case 1:  src = wzo; off = OFF_ZO;   RT = 128; KT = 32; R = 2048; K = 1024; break;
    case 2:  src = whh; off = OFF_HH;   RT = 256; KT = 8;  R = 4096; K = 256;  break;
    case 3:  src = wup; off = OFF_UP;   RT = 171; KT = 32; R = 2730; K = 1024; break;
    default: src = wdn; off = OFF_DOWN; RT = 64;  KT = 43; R = 1024; K = 1365; break;
  }
  long total = (long)RT * KT * 512;
  for (long e = (long)blockIdx.x * blockDim.x + threadIdx.x; e < total;
       e += (long)gridDim.x * blockDim.x) {
    int tile = (int)(e >> 9), w = (int)(e & 511);
    int l = w >> 4, i = w & 15;
    int tr = tile / KT, tk = tile % KT;
    int half = l >> 4, mm = l & 15;
    int v = i >> 1, lohi = i & 1;
    int kin = (v < 4 ? 2 * v : 16 + 2 * (v - 4)) + lohi + half * 8;
    int row = tr * 16 + mm, kk = tk * 32 + kin;
    float val = (row < R && kk < K) ? src[(long)row * K + kk] : 0.f;
    ws[off + (unsigned)e] = f2bf(val);
  }
}

// ---------------------------------------------------------------------------
// Persistent sLSTM kernel: 1 block x 1024 threads (32 wave32s).
// Thread t owns batch b = t/128 and channels d = (t%128)*8 .. +8.
// ---------------------------------------------------------------------------
__launch_bounds__(1024, 1)
__global__ void slstm_kernel(const float* __restrict__ x,
                             const float* __restrict__ ln_w,
                             const float* __restrict__ ln_b,
                             const float* __restrict__ conv_w,
                             const float* __restrict__ conv_b,
                             const float* __restrict__ gate_b,
                             const float* __restrict__ gn_w,
                             const float* __restrict__ gn_b,
                             const float* __restrict__ up_b,
                             const float* __restrict__ down_b,
                             const unsigned short* __restrict__ ws,
                             float* __restrict__ out) {
  extern __shared__ char smem[];
  unsigned short* s_xnb  = (unsigned short*)smem;        // [8][1024] xn, then zn
  unsigned short* s_ifb  = s_xnb + 8 * 1024;             // [8][1024] swish(conv)
  unsigned short* s_hb   = s_ifb + 8 * 1024;             // [8][1024] h state
  float*          s_lin  = (float*)(s_hb + 8 * 1024);    // [8][4096] gates
  float*          s_u    = s_lin;                        // alias: [8][2736] up out
  unsigned short* s_actb = (unsigned short*)(s_u + 8 * 2736); // [8][1376]
  float*          s_red  = (float*)((char*)s_lin + 8 * 4096 * 4); // 128 floats
  unsigned short* s_zero = (unsigned short*)(s_red + 128);        // 64B zeros

  const int tid   = threadIdx.x;
  const int wave  = tid >> 5;
  const int lane  = tid & 31;
  const int b     = tid >> 7;
  const int dBase = (tid & 127) * 8;
  const int head  = dBase >> 8;
  const v8f ZZ = {};

  for (int i = tid; i < 8 * 1024; i += 1024) s_hb[i] = 0;
  if (tid < 16) ((unsigned*)s_zero)[tid] = 0u;
  float c_s[8], n_s[8], m_s[8], r0[8], r1[8], r2[8];
#pragma unroll
  for (int e = 0; e < 8; ++e) { c_s[e]=0.f; n_s[e]=0.f; m_s[e]=0.f; r0[e]=0.f; r1[e]=0.f; r2[e]=0.f; }
  __syncthreads();

  for (int t = 0; t < TT; ++t) {
    //================ Phase A: LayerNorm + causal conv (K=4) + swish ========
    const float* xt = x + ((size_t)b * TT + t) * DD + dBase;
    float xv[8], s1 = 0.f, s2 = 0.f;
#pragma unroll
    for (int e = 0; e < 8; ++e) { float v = xt[e]; xv[e] = v; s1 += v; s2 += v * v; }
#pragma unroll
    for (int o = 16; o; o >>= 1) { s1 += __shfl_xor(s1, o, 32); s2 += __shfl_xor(s2, o, 32); }
    if (lane == 0) { int wb = (tid >> 5) & 3; s_red[b * 8 + wb * 2] = s1; s_red[b * 8 + wb * 2 + 1] = s2; }
    __syncthreads();
    if (tid < 8) {
      float a1 = 0.f, a2 = 0.f;
      for (int wv = 0; wv < 4; ++wv) { a1 += s_red[tid * 8 + wv * 2]; a2 += s_red[tid * 8 + wv * 2 + 1]; }
      float mu  = a1 * (1.f / DD);
      float var = a2 * (1.f / DD) - mu * mu;
      s_red[64 + tid * 2]     = mu;
      s_red[64 + tid * 2 + 1] = rsqrtf(var + EPSF);
    }
    __syncthreads();
    {
      float mu = s_red[64 + b * 2], rstd = s_red[64 + b * 2 + 1];
#pragma unroll
      for (int e = 0; e < 8; ++e) {
        int d = dBase + e;
        float xn = (xv[e] - mu) * rstd * ln_w[d] + ln_b[d];
        s_xnb[b * 1024 + d] = f2bf(xn);
        float zc = conv_w[d * 4 + 0] * r0[e] + conv_w[d * 4 + 1] * r1[e] +
                   conv_w[d * 4 + 2] * r2[e] + conv_w[d * 4 + 3] * xn + conv_b[d];
        r0[e] = r1[e]; r1[e] = r2[e]; r2[e] = xn;
        float sw = zc / (1.f + expf(-zc));             // swish
        s_ifb[b * 1024 + d] = f2bf(sw);
      }
    }
    __syncthreads();

    //================ Phase B: gate GEMM (if|zo + recurrent hh) =============
    // 8 lin-space row tiles per wave, processed as 2 groups of 4 to keep
    // accumulator pressure at 32 VGPRs (no spills).
    {
      int rt0 = wave * 8;                  // lin-space row tile (4096 rows)
      int h = rt0 >> 6, sub = rt0 & 63;
      bool isIf = sub < 32;                // cols 0..511 from if, 512..1023 zo
      int trL = h * 32 + (isIf ? sub : sub - 32);
      const unsigned short* srcB = isIf ? s_ifb : s_xnb;
      const unsigned short* wA   = ws + (isIf ? OFF_IF : OFF_ZO);
      int nn = lane & 15, half = lane >> 4;
#pragma unroll
      for (int g = 0; g < 2; ++g) {
        v8f acc[4];
#pragma unroll
        for (int j = 0; j < 4; ++j) acc[j] = ZZ;
        int tb = trL + g * 4;
        int rb0 = rt0 + g * 4;
        for (int kc = 0; kc < 32; ++kc) {
          v16bf bf = build_bfrag(srcB, 1024, kc * 32, s_zero);
          const unsigned short* base = wA + ((unsigned)tb * 32 + kc) * 512;
          __builtin_prefetch(base + 512, 0, 1);
          v16bf f0 = load_afrag(base);
          v16bf f1 = load_afrag(base + 32u * 512u);
          v16bf f2 = load_afrag(base + 64u * 512u);
          v16bf f3 = load_afrag(base + 96u * 512u);
          acc[0] = WMMA_BF16(f0, bf, acc[0]);
          acc[1] = WMMA_BF16(f1, bf, acc[1]);
          acc[2] = WMMA_BF16(f2, bf, acc[2]);
          acc[3] = WMMA_BF16(f3, bf, acc[3]);
        }
        for (int kc = 0; kc < 8; ++kc) {   // recurrent W_hh (K=256, head h)
          v16bf bf = build_bfrag(s_hb + h * 256, 1024, kc * 32, s_zero);
          const unsigned short* base = ws + OFF_HH + ((unsigned)rb0 * 8 + kc) * 512;
          v16bf f0 = load_afrag(base);
          v16bf f1 = load_afrag(base + 8u * 512u);
          v16bf f2 = load_afrag(base + 16u * 512u);
          v16bf f3 = load_afrag(base + 24u * 512u);
          acc[0] = WMMA_BF16(f0, bf, acc[0]);
          acc[1] = WMMA_BF16(f1, bf, acc[1]);
          acc[2] = WMMA_BF16(f2, bf, acc[2]);
          acc[3] = WMMA_BF16(f3, bf, acc[3]);
        }
        if (nn < 8) {
#pragma unroll
          for (int j = 0; j < 4; ++j) {
            int rbase = (rb0 + j) * 16 + half * 8;
            const float4 gb0 = *(const float4*)(gate_b + rbase);
            const float4 gb1 = *(const float4*)(gate_b + rbase + 4);
            float4 o0, o1;
            o0.x = acc[j][0] + gb0.x; o0.y = acc[j][1] + gb0.y;
            o0.z = acc[j][2] + gb0.z; o0.w = acc[j][3] + gb0.w;
            o1.x = acc[j][4] + gb1.x; o1.y = acc[j][5] + gb1.y;
            o1.z = acc[j][6] + gb1.z; o1.w = acc[j][7] + gb1.w;
            *(float4*)(s_lin + nn * 4096 + rbase)     = o0;
            *(float4*)(s_lin + nn * 4096 + rbase + 4) = o1;
          }
        }
      }
    }
    __syncthreads();

    //================ Phase C: sLSTM gating + GroupNorm =====================
    {
      float hv[8];
      int idx0 = dBase & 255;
      const float* lb = s_lin + b * 4096 + head * 1024;
      float gs = 0.f, gs2 = 0.f;
#pragma unroll
      for (int e = 0; e < 8; ++e) {
        float ig = lb[idx0 + e];
        float fg = lb[256 + idx0 + e];
        float zg = tanhf(lb[512 + idx0 + e]);
        float og = 1.f / (1.f + expf(-lb[768 + idx0 + e]));
        float mn = fmaxf(fg + m_s[e], ig);
        float ie = expf(ig - mn);
        float fe = expf(fg + m_s[e] - mn);
        c_s[e] = fe * c_s[e] + ie * zg;
        n_s[e] = fe * n_s[e] + ie;
        m_s[e] = mn;
        float hh = og * (c_s[e] / n_s[e]);
        hv[e] = hh; gs += hh; gs2 += hh * hh;
      }
#pragma unroll
      for (int o = 16; o; o >>= 1) { gs += __shfl_xor(gs, o, 32); gs2 += __shfl_xor(gs2, o, 32); }
      float gm = gs * (1.f / HS);
      float gr = rsqrtf(gs2 * (1.f / HS) - gm * gm + EPSF);
#pragma unroll
      for (int e = 0; e < 8; ++e) {
        int d = dBase + e;
        float zn = (hv[e] - gm) * gr * gn_w[d] + gn_b[d];
        s_xnb[b * 1024 + d] = f2bf(zn);    // reuse xn buffer for zn
        s_hb[b * 1024 + d]  = f2bf(hv[e]); // h for next step's W_hh GEMM
      }
    }
    __syncthreads();

    //================ Phase D: up GEMM (2736 x 1024 x 8) ====================
    // kc-outer / tile-inner: B fragment built once per kc (6x less DS traffic)
    {
      v8f acc[6];
#pragma unroll
      for (int j = 0; j < 6; ++j) acc[j] = ZZ;
      const int nt = (wave < 11) ? 6 : 5;          // wave-uniform tile count
      for (int kc = 0; kc < 32; ++kc) {
        v16bf bf = build_bfrag(s_xnb, 1024, kc * 32, s_zero);  // zn
#pragma unroll
        for (int ji = 0; ji < 6; ++ji) {
          if (ji < nt) {
            int rt = wave + 32 * ji;
            v16bf af = load_afrag(ws + OFF_UP + ((unsigned)rt * 32 + kc) * 512);
            acc[ji] = WMMA_BF16(af, bf, acc[ji]);
          }
        }
      }
      int nn = lane & 15, half = lane >> 4;
      if (nn < 8) {
        for (int ji = 0; ji < nt; ++ji) {
          int rt = wave + 32 * ji;
#pragma unroll
          for (int vv = 0; vv < 8; ++vv) {
            int row = rt * 16 + half * 8 + vv;
            if (row < 2 * UU) s_u[nn * 2736 + row] = acc[ji][vv] + up_b[row];
          }
        }
      }
    }
    __syncthreads();

    //================ Phase E: gelu(u1) * u2 -> bf16 (K padded to 1376) =====
    for (int idx = tid; idx < 8 * 1376; idx += 1024) {
      int bb = idx / 1376, j = idx - bb * 1376;
      float v = 0.f;
      if (j < UU) {
        float u1 = s_u[bb * 2736 + j];
        float u2 = s_u[bb * 2736 + UU + j];
        float g = 0.5f * u1 * (1.f + tanhf(0.7978845608f * (u1 + 0.044715f * u1 * u1 * u1)));
        v = g * u2;
      }
      s_actb[bb * 1376 + j] = f2bf(v);
    }
    __syncthreads();

    //================ Phase F: down GEMM (1024 x 1376 x 8) + residual =======
    {
      v8f acc[2]; acc[0] = ZZ; acc[1] = ZZ;
      for (int kc = 0; kc < 43; ++kc) {
        v16bf bf = build_bfrag(s_actb, 1376, kc * 32, s_zero);
#pragma unroll
        for (int ji = 0; ji < 2; ++ji) {
          int rt = wave + 32 * ji;
          v16bf af = load_afrag(ws + OFF_DOWN + ((unsigned)rt * 43 + kc) * 512);
          acc[ji] = WMMA_BF16(af, bf, acc[ji]);
        }
      }
      int nn = lane & 15, half = lane >> 4;
      if (nn < 8) {
#pragma unroll
        for (int ji = 0; ji < 2; ++ji) {
          int rt = wave + 32 * ji;
          int row = rt * 16 + half * 8;
          size_t o = ((size_t)nn * TT + t) * DD + row;
          const float4 xa  = *(const float4*)(x + o);
          const float4 xb  = *(const float4*)(x + o + 4);
          const float4 db0 = *(const float4*)(down_b + row);
          const float4 db1 = *(const float4*)(down_b + row + 4);
          float4 o0, o1;
          o0.x = acc[ji][0] + db0.x + xa.x; o0.y = acc[ji][1] + db0.y + xa.y;
          o0.z = acc[ji][2] + db0.z + xa.z; o0.w = acc[ji][3] + db0.w + xa.w;
          o1.x = acc[ji][4] + db1.x + xb.x; o1.y = acc[ji][5] + db1.y + xb.y;
          o1.z = acc[ji][6] + db1.z + xb.z; o1.w = acc[ji][7] + db1.w + xb.w;
          *(float4*)(out + o)     = o0;
          *(float4*)(out + o + 4) = o1;
        }
      }
    }
    __syncthreads();
  }
}

// ---------------------------------------------------------------------------
extern "C" void kernel_launch(void* const* d_in, const int* in_sizes, int n_in,
                              void* d_out, int out_size, void* d_ws, size_t ws_size,
                              hipStream_t stream) {
  const float* x      = (const float*)d_in[0];
  const float* ln_w   = (const float*)d_in[1];
  const float* ln_b   = (const float*)d_in[2];
  const float* conv_w = (const float*)d_in[3];
  const float* conv_b = (const float*)d_in[4];
  const float* w_if   = (const float*)d_in[5];
  const float* w_zo   = (const float*)d_in[6];
  const float* w_hh   = (const float*)d_in[7];
  const float* bias   = (const float*)d_in[8];
  const float* gn_w   = (const float*)d_in[9];
  const float* gn_b   = (const float*)d_in[10];
  const float* up_w   = (const float*)d_in[11];
  const float* up_b   = (const float*)d_in[12];
  const float* dn_w   = (const float*)d_in[13];
  const float* dn_b   = (const float*)d_in[14];
  unsigned short* ws  = (unsigned short*)d_ws;   // needs ~18.9 MB

  // 1) convert + swizzle all weights fp32 -> bf16 A-fragments (once per call)
  dim3 pg(512, 5);
  pack_weights_kernel<<<pg, 256, 0, stream>>>(w_if, w_zo, w_hh, up_w, dn_w, ws);

  // 2) persistent recurrent kernel: one 32-wave workgroup, all 8 batches
  size_t smem = (size_t)3 * 8 * 1024 * 2   // xn/if/h bf16 buffers
              + (size_t)8 * 4096 * 4       // lin (aliased with u + actb)
              + 512                         // reductions
              + 64;                         // zero region for padded B columns
  slstm_kernel<<<1, 1024, smem, stream>>>(x, ln_w, ln_b, conv_w, conv_b, bias,
                                          gn_w, gn_b, up_b, dn_b, ws,
                                          (float*)d_out);
}
